// TarMAC_29214367547983
// MI455X (gfx1250) — matrix operand order repaired and checked by reading, not compiled, not run on hardware
//
#include <hip/hip_runtime.h>
#include <hip/hip_bf16.h>
#include <math.h>

#define NN 50000
#define NE 800000
#define HID 128
#define MSGD 128
#define KEYD 32
#define TWOH 256
#define VSQD 192   // 128 val + 32 sig + 32 que
#define G3 384

typedef _Float16 v16h __attribute__((ext_vector_type(16)));
typedef _Float16 h8   __attribute__((ext_vector_type(8)));
typedef float    v8f  __attribute__((ext_vector_type(8)));

// ---------- helpers ----------
__device__ __forceinline__ float sigm(float x) { return 1.0f / (1.0f + __expf(-x)); }

// order-preserving float <-> uint mapping for atomicMax
__device__ __forceinline__ unsigned fkey(float f) {
    unsigned u = __float_as_uint(f);
    return (u & 0x80000000u) ? ~u : (u | 0x80000000u);
}
__device__ __forceinline__ float funkey(unsigned k) {
    unsigned u = (k & 0x80000000u) ? (k & 0x7fffffffu) : ~k;
    return __uint_as_float(u);
}

// ---------- prep kernels ----------
__global__ void conv_f32_to_f16(const float* __restrict__ s, _Float16* __restrict__ d, int n) {
    int i = blockIdx.x * blockDim.x + threadIdx.x;
    if (i < n) d[i] = (_Float16)s[i];
}

// Wcat16[n][k] (row-major [192,256]) = transposed concat of W_val/W_sig/W_que ([256,*])
__global__ void build_wcat(const float* __restrict__ Wv, const float* __restrict__ Ws,
                           const float* __restrict__ Wq, _Float16* __restrict__ out) {
    int i = blockIdx.x * blockDim.x + threadIdx.x;   // over 192*256
    if (i >= VSQD * TWOH) return;
    int n = i / TWOH, k = i % TWOH;
    float w;
    if (n < 128)      w = Wv[k * 128 + n];
    else if (n < 160) w = Ws[k * 32 + (n - 128)];
    else              w = Wq[k * 32 + (n - 160)];
    out[n * TWOH + k] = (_Float16)w;
}

__global__ void build_bcat(const float* __restrict__ bv, const float* __restrict__ bs,
                           const float* __restrict__ bq, float* __restrict__ out) {
    int i = blockIdx.x * blockDim.x + threadIdx.x;
    if (i >= VSQD) return;
    out[i] = (i < 128) ? bv[i] : (i < 160) ? bs[i - 128] : bq[i - 160];
}

// dst16[i*256 + j] = j<128 ? a[i*128+j] : b[i*128+(j-128)]   (f32 -> f16 pack)
__global__ void pack2_f16(const float* __restrict__ a, const float* __restrict__ b,
                          _Float16* __restrict__ d) {
    int i = blockIdx.x * blockDim.x + threadIdx.x;   // over NN*256
    if (i >= NN * TWOH) return;
    int node = i >> 8, j = i & 255;
    float v = (j < 128) ? a[node * 128 + j] : b[node * 128 + (j - 128)];
    d[i] = (_Float16)v;
}

// ---------- WMMA GEMM: C[M,N] = A[M,K] @ Bt[N,K]^T + bias[N] ----------
// A f16 row-major (lda halfs), Bt f16 row-major (ldb=K halfs, rows are output cols),
// C f32 row-major (ldc). M = Mtiles*16, N = Ntiles*16, K % 32 == 0.
__global__ void gemm_wmma_f16(const _Float16* __restrict__ A, int lda,
                              const _Float16* __restrict__ Bt, int ldb,
                              const float* __restrict__ bias,
                              float* __restrict__ C, int ldc,
                              int Mtiles, int Ntiles, int K) {
    int wave = blockIdx.x * (blockDim.x >> 5) + (threadIdx.x >> 5);
    int lane = threadIdx.x & 31;
    int total = Mtiles * Ntiles;
    if (wave >= total) return;               // wave-uniform: EXEC stays all-ones for WMMA
    int nt = wave % Ntiles;
    int mt = wave / Ntiles;
    int l15 = lane & 15;
    int khi = lane >> 4;

    // A fragment rows: lane l holds row m = l&15; K pattern: kb + 8*khi + [0..7] and kb+16+8*khi+[0..7]
    const _Float16* ap = A + (size_t)(mt * 16 + l15) * lda + 8 * khi;
    // B fragment: lane l holds output column n = l&15; K pattern: kb + 8*khi + [0..7] and kb+16+8*khi+[0..7]
    const _Float16* bp = Bt + (size_t)(nt * 16 + l15) * ldb + 8 * khi;

    v8f acc = {};
    for (int kb = 0; kb < K; kb += 32) {
        h8 a0 = *(const h8*)(ap + kb);
        h8 a1 = *(const h8*)(ap + kb + 16);
        h8 b0 = *(const h8*)(bp + kb);
        h8 b1 = *(const h8*)(bp + kb + 16);
        v16h av = __builtin_shufflevector(a0, a1, 0,1,2,3,4,5,6,7,8,9,10,11,12,13,14,15);
        v16h bv = __builtin_shufflevector(b0, b1, 0,1,2,3,4,5,6,7,8,9,10,11,12,13,14,15);
        acc = __builtin_amdgcn_wmma_f32_16x16x32_f16(false, av, false, bv,
                                                     (short)0, acc, false, false);
    }
    int ccol = nt * 16 + l15;
    float bvs = bias ? bias[ccol] : 0.0f;
    int crow0 = mt * 16 + 8 * khi;           // C layout: lane group khi holds rows 8*khi..8*khi+7
    #pragma unroll
    for (int i = 0; i < 8; i++)
        C[(size_t)(crow0 + i) * ldc + ccol] = acc[i] + bvs;
}

// ---------- per-round edge kernels ----------
__global__ void init_round(float* __restrict__ c, float* __restrict__ denom,
                           unsigned* __restrict__ emax) {
    int i = blockIdx.x * blockDim.x + threadIdx.x;   // over NN*128
    if (i >= NN * MSGD) return;
    c[i] = 0.0f;
    if (i < NN) { denom[i] = 0.0f; emax[i] = 0u; }   // key 0 == most negative
}

// wave per edge: e = dot(s[src], q[dst]) / KEY ; atomicMax into emax[dst]
__global__ void edge_score(const float* __restrict__ vsq, const int* __restrict__ src,
                           const int* __restrict__ dst, float* __restrict__ e,
                           unsigned* __restrict__ emax) {
    int edge = blockIdx.x * (blockDim.x >> 5) + (threadIdx.x >> 5);
    int lane = threadIdx.x & 31;
    if (edge >= NE) return;
    int s = src[edge], d = dst[edge];
    float p = vsq[(size_t)s * VSQD + 128 + lane] * vsq[(size_t)d * VSQD + 160 + lane];
    #pragma unroll
    for (int off = 16; off > 0; off >>= 1) p += __shfl_xor(p, off, 32);
    if (lane == 0) {
        float sc = p * (1.0f / (float)KEYD);
        e[edge] = sc;
        atomicMax(&emax[d], fkey(sc));
    }
}

__global__ void edge_exp(const float* __restrict__ e, const int* __restrict__ dst,
                         const unsigned* __restrict__ emax, float* __restrict__ ee,
                         float* __restrict__ denom) {
    int i = blockIdx.x * blockDim.x + threadIdx.x;
    if (i >= NE) return;
    int d = dst[i];
    float v = __expf(e[i] - funkey(emax[d]));
    ee[i] = v;
    atomicAdd(&denom[d], v);
}

// wave per edge: c[dst] += (ee/denom[dst]) * v[src]
__global__ void edge_agg(const float* __restrict__ vsq, const int* __restrict__ src,
                         const int* __restrict__ dst, const float* __restrict__ ee,
                         const float* __restrict__ denom, float* __restrict__ c) {
    int edge = blockIdx.x * (blockDim.x >> 5) + (threadIdx.x >> 5);
    int lane = threadIdx.x & 31;
    if (edge >= NE) return;
    int s = src[edge], d = dst[edge];
    float a = ee[edge] / denom[d];
    const float* vp = vsq + (size_t)s * VSQD;
    float* cp = c + (size_t)d * MSGD;
    #pragma unroll
    for (int t = 0; t < 4; t++) {
        int j = lane + 32 * t;
        atomicAdd(&cp[j], a * vp[j]);
    }
}

// ---------- GRU elementwise ----------
__global__ void gru_elem(const float* __restrict__ gi, const float* __restrict__ gh,
                         const float* __restrict__ hprev, float* __restrict__ hout) {
    int idx = blockIdx.x * blockDim.x + threadIdx.x;   // over NN*128
    if (idx >= NN * HID) return;
    int i = idx >> 7, j = idx & 127;
    const float* gii = gi + (size_t)i * G3;
    const float* ghh = gh + (size_t)i * G3;
    float r = sigm(gii[j]       + ghh[j]);
    float z = sigm(gii[128 + j] + ghh[128 + j]);
    float n = tanhf(gii[256 + j] + r * ghh[256 + j]);
    hout[idx] = (1.0f - z) * n + z * hprev[idx];
}

// ---------- launch ----------
extern "C" void kernel_launch(void* const* d_in, const int* in_sizes, int n_in,
                              void* d_out, int out_size, void* d_ws, size_t ws_size,
                              hipStream_t stream) {
    const float* x    = (const float*)d_in[0];
    const float* h0   = (const float*)d_in[1];
    const int*   src  = (const int*)  d_in[2];
    const int*   dst  = (const int*)  d_in[3];
    const float* Wval = (const float*)d_in[4];
    const float* bval = (const float*)d_in[5];
    const float* Wsig = (const float*)d_in[6];
    const float* bsig = (const float*)d_in[7];
    const float* Wque = (const float*)d_in[8];
    const float* bque = (const float*)d_in[9];
    const float* Wih  = (const float*)d_in[10];
    const float* bih  = (const float*)d_in[11];
    const float* Whh  = (const float*)d_in[12];
    const float* bhh  = (const float*)d_in[13];
    float* hout = (float*)d_out;

    char* ws = (char*)d_ws;
    size_t off = 0;
    auto alloc = [&](size_t bytes) -> void* {
        void* p = ws + off;
        off += (bytes + 255) & ~(size_t)255;
        return p;
    };
    _Float16* xh16   = (_Float16*)alloc((size_t)NN * TWOH * 2);
    _Float16* xc16   = (_Float16*)alloc((size_t)NN * TWOH * 2);
    float*    vsq    = (float*)   alloc((size_t)NN * VSQD * 4);
    float*    cbuf   = (float*)   alloc((size_t)NN * MSGD * 4);
    float*    gi     = (float*)   alloc((size_t)NN * G3 * 4);
    float*    gh     = (float*)   alloc((size_t)NN * G3 * 4);
    float*    h1     = (float*)   alloc((size_t)NN * HID * 4);
    float*    ebuf   = (float*)   alloc((size_t)NE * 4);
    float*    eebuf  = (float*)   alloc((size_t)NE * 4);
    unsigned* emax   = (unsigned*)alloc((size_t)NN * 4);
    float*    denom  = (float*)   alloc((size_t)NN * 4);
    _Float16* wcat16 = (_Float16*)alloc((size_t)VSQD * TWOH * 2);
    _Float16* wih16  = (_Float16*)alloc((size_t)G3 * TWOH * 2);
    _Float16* whh16  = (_Float16*)alloc((size_t)G3 * HID * 2);
    float*    bcat   = (float*)   alloc((size_t)VSQD * 4);

    const int T = 256;
    // one-time weight prep (deterministic each call)
    build_wcat<<<(VSQD * TWOH + T - 1) / T, T, 0, stream>>>(Wval, Wsig, Wque, wcat16);
    build_bcat<<<1, T, 0, stream>>>(bval, bsig, bque, bcat);
    conv_f32_to_f16<<<(G3 * TWOH + T - 1) / T, T, 0, stream>>>(Wih, wih16, G3 * TWOH);
    conv_f32_to_f16<<<(G3 * HID + T - 1) / T, T, 0, stream>>>(Whh, whh16, G3 * HID);

    const int Mt = NN / 16;                       // 3125 exactly
    const int wavesPerBlk = T / 32;               // 8
    auto gemmBlocks = [&](int tiles) { return (tiles + wavesPerBlk - 1) / wavesPerBlk; };

    for (int round = 0; round < 2; round++) {
        const float* hcur = (round == 0) ? h0 : h1;
        float* hnext      = (round == 0) ? h1 : hout;

        // inp = [x | h]  (f16)
        pack2_f16<<<(NN * TWOH + T - 1) / T, T, 0, stream>>>(x, hcur, xh16);

        // vsq = inp @ [Wval|Wsig|Wque] + bias   [NN,192]
        gemm_wmma_f16<<<gemmBlocks(Mt * (VSQD / 16)), T, 0, stream>>>(
            xh16, TWOH, wcat16, TWOH, bcat, vsq, VSQD, Mt, VSQD / 16, TWOH);

        // segment softmax attention + aggregation
        init_round<<<(NN * MSGD + T - 1) / T, T, 0, stream>>>(cbuf, denom, emax);
        edge_score<<<(NE + wavesPerBlk - 1) / wavesPerBlk, T, 0, stream>>>(vsq, src, dst, ebuf, emax);
        edge_exp<<<(NE + T - 1) / T, T, 0, stream>>>(ebuf, dst, emax, eebuf, denom);
        edge_agg<<<(NE + wavesPerBlk - 1) / wavesPerBlk, T, 0, stream>>>(vsq, src, dst, eebuf, denom, cbuf);

        // GRU: gi = [x|c] @ Wih^T + bih ; gh = h @ Whh^T + bhh
        pack2_f16<<<(NN * TWOH + T - 1) / T, T, 0, stream>>>(x, cbuf, xc16);
        gemm_wmma_f16<<<gemmBlocks(Mt * (G3 / 16)), T, 0, stream>>>(
            xc16, TWOH, wih16, TWOH, bih, gi, G3, Mt, G3 / 16, TWOH);
        gemm_wmma_f16<<<gemmBlocks(Mt * (G3 / 16)), T, 0, stream>>>(
            xh16 + 128, TWOH, whh16, HID, bhh, gh, G3, Mt, G3 / 16, HID);
        gru_elem<<<(NN * HID + T - 1) / T, T, 0, stream>>>(gi, gh, hcur, hnext);
    }
}
